// PFNet_39857296507464
// MI455X (gfx1250) — matrix operand order repaired and checked by reading, not compiled, not run on hardware
//
#include <hip/hip_runtime.h>
#include <hip/hip_bf16.h>

// ---------------- problem constants ----------------
#define BSZ      8
#define NPTS     5000
#define FDIM     15
#define NCLS_IN  12
#define NCLS_OUT 8
#define NMOM     3
#define HDIM     256
#define BIN_SIZE 500
#define N_BINS   10
#define K_NN     16
#define ENC_DIM  26      // 12 one-hot + 14 props
#define ENC_PAD  32
#define XR_DIM   34      // 26 + 8 logits
#define XR_PAD   64
#define R_LD     100     // MAX_NUM_BINS/2
#define BN       (BSZ*NPTS)                 // 40000
#define NEDGE    (BSZ*N_BINS*BIN_SIZE*K_NN) // 640000

#define ACT_NONE 0
#define ACT_ELU  1
#define ACT_SELU 2

// LDS tile strides (ushorts). 40*2 = 80 bytes: multiple of 16 -> uint4 aligned.
#define ASTR 40
#define BSTR 40

typedef __attribute__((ext_vector_type(16))) __bf16          v16bf;
typedef __attribute__((ext_vector_type(8)))  float           v8f;

struct Frag32 { uint4 lo, hi; };   // 32 bytes == v16bf

static __device__ __forceinline__ unsigned int f2bf(float f) {
    unsigned int u = __float_as_uint(f);
    u += 0x7FFFu + ((u >> 16) & 1u);      // round-to-nearest-even
    return u >> 16;
}
static __device__ __forceinline__ unsigned int pack2(float a, float b) {
    return f2bf(a) | (f2bf(b) << 16);
}
static __device__ __forceinline__ float act_selu(float x) {
    const float a = 1.6732632423543772f, s = 1.0507009873554805f;
    return s * (x > 0.0f ? x : a * (__expf(x) - 1.0f));
}
static __device__ __forceinline__ v16bf ld_frag(const unsigned short* p) {
    Frag32 f;
    f.lo = *(const uint4*)(p);
    f.hi = *(const uint4*)(p + 8);
    return __builtin_bit_cast(v16bf, f);
}

// =====================================================================
// WMMA bf16 GEMM: C = act( rowscale * (A[M,K] @ W[K,Nc] + bias) )
// 128 threads (4 waves), block tile 128x64, K-step 32.
// Each wave: 32 rows x 64 cols = 8 wmma accumulators.
// A tile row-major [128][ASTR]; W tile transposed [64][BSTR] so every
// fragment is two contiguous 16B LDS reads (ds_load_b128).
// =====================================================================
__global__ __launch_bounds__(128) void gemm_bf16_kernel(
    const float* __restrict__ A, int lda, int K,
    const float* __restrict__ W, int ldw,
    const float* __restrict__ bias,
    const float* __restrict__ rowscale,
    float* __restrict__ C, int ldc,
    int M, int Nc, int act)
{
    __shared__ unsigned short As[128][ASTR];
    __shared__ unsigned short WsT[64][BSTR];

    const int tid  = threadIdx.x;
    const int wave = tid >> 5, lane = tid & 31;
    const int m16  = lane & 15, hi = lane >> 4;
    const int r0   = blockIdx.x * 128;
    const int c0   = blockIdx.y * 64;

    v8f acc[2][4];
    #pragma unroll
    for (int s = 0; s < 2; ++s)
        #pragma unroll
        for (int t = 0; t < 4; ++t) acc[s][t] = v8f{};

    for (int kb = 0; kb < K; kb += 32) {
        // ---- stage A tile: 128 rows x 32 k ----
        if (kb + 32 <= K && r0 + 128 <= M) {
            #pragma unroll
            for (int it = 0; it < 8; ++it) {            // 1024 quads / 128 thr
                int t = tid + it * 128;
                int mm = t >> 3, q = t & 7;
                const float4 f = *(const float4*)&A[(size_t)(r0 + mm) * lda + kb + q * 4];
                uint2 pk; pk.x = pack2(f.x, f.y); pk.y = pack2(f.z, f.w);
                *(uint2*)&As[mm][q * 4] = pk;
            }
        } else {
            for (int t = tid; t < 128 * 32; t += 128) {
                int mm = t >> 5, kk = t & 31;
                int gm = r0 + mm, gk = kb + kk;
                float v = (gm < M && gk < K) ? A[(size_t)gm * lda + gk] : 0.0f;
                As[mm][kk] = (unsigned short)f2bf(v);
            }
        }
        // ---- stage W tile transposed: 64 cols x 32 k ----
        if (kb + 32 <= K && c0 + 64 <= Nc) {
            #pragma unroll
            for (int it = 0; it < 4; ++it) {            // 512 quads / 128 thr
                int t = tid + it * 128;
                int cc = t >> 3, q = t & 7;
                const float* wp = &W[(size_t)(kb + q * 4) * ldw + c0 + cc];
                float w0 = wp[0];
                float w1 = wp[ldw];
                float w2 = wp[2 * (size_t)ldw];
                float w3 = wp[3 * (size_t)ldw];
                uint2 pk; pk.x = pack2(w0, w1); pk.y = pack2(w2, w3);
                *(uint2*)&WsT[cc][q * 4] = pk;
            }
        } else {
            for (int t = tid; t < 64 * 32; t += 128) {
                int cc = t >> 5, kk = t & 31;
                int gk = kb + kk, gc = c0 + cc;
                float v = (gk < K && gc < Nc) ? W[(size_t)gk * ldw + gc] : 0.0f;
                WsT[cc][kk] = (unsigned short)f2bf(v);
            }
        }
        __syncthreads();

        // ---- fragments: 2x b128 per fragment ----
        v16bf af[2];
        #pragma unroll
        for (int s = 0; s < 2; ++s) {
            const unsigned short* ap = &As[wave * 32 + s * 16 + m16][hi * 16];
            Frag32 f;
            f.lo = *(const uint4*)(ap);
            f.hi = *(const uint4*)(ap + 8);             // K = hi*16 + 8 ..
            af[s] = __builtin_bit_cast(v16bf, f);
        }
        #pragma unroll
        for (int t = 0; t < 4; ++t) {
            v16bf bf = ld_frag(&WsT[t * 16 + m16][hi * 16]);
            #pragma unroll
            for (int s = 0; s < 2; ++s)
                acc[s][t] = __builtin_amdgcn_wmma_f32_16x16x32_bf16(
                    false, af[s], false, bf, (short)0, acc[s][t], false, false);
        }
        __syncthreads();
    }

    // ---- epilogue ----
    #pragma unroll
    for (int t = 0; t < 4; ++t) {
        int col = c0 + t * 16 + m16;
        if (col >= Nc) continue;
        float bv = bias ? bias[col] : 0.0f;
        #pragma unroll
        for (int s = 0; s < 2; ++s) {
            int rbase = r0 + wave * 32 + s * 16 + hi * 8;
            #pragma unroll
            for (int g = 0; g < 8; ++g) {
                int r = rbase + g;
                if (r >= M) continue;
                float x = acc[s][t][g] + bv;
                if (rowscale) x *= rowscale[r];
                if (act == ACT_ELU)  x = x > 0.0f ? x : (__expf(x) - 1.0f);
                else if (act == ACT_SELU) x = act_selu(x);
                C[(size_t)r * ldc + col] = x;
            }
        }
    }
}

// =====================================================================
// Distance Gram GEMM per bin:  G[bin] = P_bin @ P_bin^T  (500x500, ld 512)
// Same tiling; B-tile source is naturally k-contiguous (P row-major).
// =====================================================================
__global__ __launch_bounds__(128) void gemm_dist_kernel(
    const float* __restrict__ parts,   // [B*N, 256], bin-contiguous
    float* __restrict__ G)             // [B*N_BINS, 500, 512]
{
    __shared__ unsigned short As[128][ASTR];
    __shared__ unsigned short WsT[64][BSTR];

    const int binid = blockIdx.z;                 // 0..79
    const float* P  = parts + (size_t)binid * BIN_SIZE * HDIM;
    float* Gb       = G + (size_t)binid * BIN_SIZE * 512;

    const int tid  = threadIdx.x;
    const int wave = tid >> 5, lane = tid & 31;
    const int m16  = lane & 15, hi = lane >> 4;
    const int r0   = blockIdx.x * 128;
    const int c0   = blockIdx.y * 64;

    v8f acc[2][4];
    #pragma unroll
    for (int s = 0; s < 2; ++s)
        #pragma unroll
        for (int t = 0; t < 4; ++t) acc[s][t] = v8f{};

    for (int kb = 0; kb < HDIM; kb += 32) {
        if (r0 + 128 <= BIN_SIZE) {
            #pragma unroll
            for (int it = 0; it < 8; ++it) {
                int t = tid + it * 128;
                int mm = t >> 3, q = t & 7;
                const float4 f = *(const float4*)&P[(size_t)(r0 + mm) * HDIM + kb + q * 4];
                uint2 pk; pk.x = pack2(f.x, f.y); pk.y = pack2(f.z, f.w);
                *(uint2*)&As[mm][q * 4] = pk;
            }
        } else {
            for (int t = tid; t < 128 * 32; t += 128) {
                int mm = t >> 5, kk = t & 31;
                int gm = r0 + mm;
                float v = (gm < BIN_SIZE) ? P[(size_t)gm * HDIM + kb + kk] : 0.0f;
                As[mm][kk] = (unsigned short)f2bf(v);
            }
        }
        #pragma unroll
        for (int it = 0; it < 4; ++it) {              // B = P^T: k-contiguous
            int t = tid + it * 128;
            int cc = t >> 3, q = t & 7;
            int gc = c0 + cc;
            uint2 pk;
            if (gc < BIN_SIZE) {
                const float4 f = *(const float4*)&P[(size_t)gc * HDIM + kb + q * 4];
                pk.x = pack2(f.x, f.y); pk.y = pack2(f.z, f.w);
            } else { pk.x = 0u; pk.y = 0u; }
            *(uint2*)&WsT[cc][q * 4] = pk;
        }
        __syncthreads();

        v16bf af[2];
        #pragma unroll
        for (int s = 0; s < 2; ++s) {
            const unsigned short* ap = &As[wave * 32 + s * 16 + m16][hi * 16];
            Frag32 f;
            f.lo = *(const uint4*)(ap);
            f.hi = *(const uint4*)(ap + 8);
            af[s] = __builtin_bit_cast(v16bf, f);
        }
        #pragma unroll
        for (int t = 0; t < 4; ++t) {
            v16bf bf = ld_frag(&WsT[t * 16 + m16][hi * 16]);
            #pragma unroll
            for (int s = 0; s < 2; ++s)
                acc[s][t] = __builtin_amdgcn_wmma_f32_16x16x32_bf16(
                    false, af[s], false, bf, (short)0, acc[s][t], false, false);
        }
        __syncthreads();
    }

    #pragma unroll
    for (int t = 0; t < 4; ++t) {
        int col = c0 + t * 16 + m16;
        #pragma unroll
        for (int s = 0; s < 2; ++s) {
            int rbase = r0 + wave * 32 + s * 16 + hi * 8;
            #pragma unroll
            for (int g = 0; g < 8; ++g) {
                int r = rbase + g;
                if (r < BIN_SIZE) Gb[(size_t)r * 512 + col] = acc[s][t][g];
            }
        }
    }
}

// ---------------- elementwise / graph kernels ----------------
__global__ __launch_bounds__(256) void encode_kernel(
    const float* __restrict__ X, float* __restrict__ enc)
{
    int node = blockIdx.x * 256 + threadIdx.x;
    if (node >= BN) return;
    const float* x = X + (size_t)node * FDIM;
    float* e = enc + (size_t)node * ENC_PAD;
    int id = (int)x[0];
    #pragma unroll
    for (int c = 0; c < NCLS_IN; ++c) e[c] = (c == id) ? 1.0f : 0.0f;
    #pragma unroll
    for (int c = 0; c < FDIM - 1; ++c) e[NCLS_IN + c] = x[1 + c];
    #pragma unroll
    for (int c = ENC_DIM; c < ENC_PAD; ++c) e[c] = 0.0f;
}

__global__ __launch_bounds__(256) void bin_kernel(
    const float* __restrict__ xd, const float* __restrict__ R,
    int* __restrict__ binidx)
{
    int node = blockIdx.x * 256 + threadIdx.x;
    if (node >= BN) return;
    const float* xr = xd + (size_t)node * HDIM;
    float mul[N_BINS / 2] = {0.f, 0.f, 0.f, 0.f, 0.f};
    for (int k = 0; k < HDIM; ++k) {
        float xv = xr[k];
        const float* rr = R + (size_t)k * R_LD;
        #pragma unroll
        for (int j = 0; j < N_BINS / 2; ++j) mul[j] += xv * rr[j];
    }
    int best = 0; float bv = mul[0];
    #pragma unroll
    for (int j = 1; j < N_BINS; ++j) {
        float v = (j < N_BINS / 2) ? mul[j] : -mul[j - N_BINS / 2];
        if (v > bv) { bv = v; best = j; }
    }
    binidx[node] = best;
}

__global__ __launch_bounds__(256) void sort_kernel(
    const int* __restrict__ binidx, int* __restrict__ order)
{
    __shared__ int hist[N_BINS];
    __shared__ int offs[N_BINS];
    const int b = blockIdx.x, tid = threadIdx.x;
    if (tid < N_BINS) hist[tid] = 0;
    __syncthreads();
    for (int n = tid; n < NPTS; n += 256) atomicAdd(&hist[binidx[b * NPTS + n]], 1);
    __syncthreads();
    if (tid == 0) { int run = 0; for (int j = 0; j < N_BINS; ++j) { offs[j] = run; run += hist[j]; } }
    __syncthreads();
    for (int n = tid; n < NPTS; n += 256) {
        int pos = atomicAdd(&offs[binidx[b * NPTS + n]], 1);
        order[b * NPTS + pos] = n;
    }
}

__global__ __launch_bounds__(256) void parts_kernel(
    const float* __restrict__ xd, const int* __restrict__ order,
    float* __restrict__ parts, float* __restrict__ sq)
{
    __shared__ float red[256];
    const int p = blockIdx.x;          // 0..BN-1 (bin-sorted row)
    const int b = p / NPTS, tid = threadIdx.x;
    const int srcn = order[p];
    float v = xd[((size_t)b * NPTS + srcn) * HDIM + tid];
    parts[(size_t)p * HDIM + tid] = v;
    red[tid] = v * v;
    __syncthreads();
    for (int s = 128; s > 0; s >>= 1) { if (tid < s) red[tid] += red[tid + s]; __syncthreads(); }
    if (tid == 0) sq[p] = red[0];
}

__global__ __launch_bounds__(128) void topk_kernel(
    const float* __restrict__ G, const float* __restrict__ sq,
    const int* __restrict__ order,
    int* __restrict__ esrc, int* __restrict__ edst, float* __restrict__ eval,
    float* __restrict__ norm)
{
    __shared__ float dmv[512];
    __shared__ float bv[128];
    __shared__ int   bi[128];
    __shared__ float selval[K_NN];
    __shared__ int   selidx[K_NN];

    const int blk = blockIdx.x, tid = threadIdx.x;
    const int b   = blk / (N_BINS * BIN_SIZE);
    const int rem = blk % (N_BINS * BIN_SIZE);
    const int bin = rem / BIN_SIZE, i = rem % BIN_SIZE;
    const int binrow0 = b * NPTS + bin * BIN_SIZE;
    const float* Gr = G + (((size_t)(b * N_BINS + bin)) * BIN_SIZE + i) * 512;
    const float sqi = sq[binrow0 + i];

    for (int j = tid; j < 512; j += 128) {
        float dm = -1e30f;
        if (j < BIN_SIZE) {
            float d2 = sqi + sq[binrow0 + j] - 2.0f * Gr[j];
            dm = __expf(-0.1f * sqrtf(fmaxf(d2, 1e-6f)));
        }
        dmv[j] = dm;
    }
    __syncthreads();

    for (int k = 0; k < K_NN; ++k) {
        float best = -1e30f; int bidx = 0;
        for (int j = tid; j < 512; j += 128) {
            float v = dmv[j];
            if (v > best) { best = v; bidx = j; }
        }
        bv[tid] = best; bi[tid] = bidx;
        __syncthreads();
        for (int s = 64; s > 0; s >>= 1) {
            if (tid < s && bv[tid + s] > bv[tid]) { bv[tid] = bv[tid + s]; bi[tid] = bi[tid + s]; }
            __syncthreads();
        }
        if (tid == 0) { selidx[k] = bi[0]; selval[k] = bv[0]; dmv[bi[0]] = -1e30f; }
        __syncthreads();
    }

    if (tid < K_NN) {
        int e    = (bin * BIN_SIZE + i) * K_NN + tid;
        int base = b * (N_BINS * BIN_SIZE * K_NN);
        esrc[base + e] = order[binrow0 + i];
        edst[base + e] = order[binrow0 + selidx[tid]];
        eval[base + e] = selval[tid];
    }
    if (tid == 0) {
        float deg = 0.0f;
        #pragma unroll
        for (int k = 0; k < K_NN; ++k) deg += fabsf(selval[k]);
        norm[b * NPTS + order[binrow0 + i]] = rsqrtf(deg + 1e-6f);
    }
}

__global__ __launch_bounds__(256) void gather_kernel(
    const int* __restrict__ esrc, const int* __restrict__ edst,
    const float* __restrict__ eval, const float* __restrict__ fpre,
    float* __restrict__ fagg)
{
    const int blk = blockIdx.x;              // b*NPTS + p
    const int b = blk / NPTS, tid = threadIdx.x;
    const int eb = b * (N_BINS * BIN_SIZE * K_NN) + (blk % NPTS) * K_NN;
    const int s = esrc[eb];
    float acc = 0.0f;
    #pragma unroll
    for (int k = 0; k < K_NN; ++k) {
        int   d = edst[eb + k];
        float v = eval[eb + k];
        acc += v * fpre[((size_t)b * NPTS + d) * HDIM + tid];
    }
    fagg[((size_t)b * NPTS + s) * HDIM + tid] = acc;
}

__global__ __launch_bounds__(256) void combine_kernel(
    const float* __restrict__ tgate, const float* __restrict__ fagg,
    const float* __restrict__ fhet, const float* __restrict__ norm,
    float* __restrict__ out)
{
    size_t idx = (size_t)blockIdx.x * 256 + threadIdx.x;
    size_t row = idx >> 8;
    float g = 1.0f / (1.0f + __expf(-tgate[idx]));
    float x = g * (fagg[idx] * norm[row]) + (1.0f - g) * fhet[idx];
    out[idx] = act_selu(x);
}

__global__ __launch_bounds__(256) void outproj_kernel(
    const float* __restrict__ Hb, const float* __restrict__ W,
    const float* __restrict__ bias, float* __restrict__ out, int nOut)
{
    int idx = blockIdx.x * 256 + threadIdx.x;
    int total = BN * nOut;
    if (idx >= total) return;
    int c = idx % nOut, node = idx / nOut;
    const float* h = Hb + (size_t)node * HDIM;
    float acc = bias[c];
    for (int k = 0; k < HDIM; ++k) acc += h[k] * W[k * nOut + c];
    out[idx] = acc;
}

__global__ __launch_bounds__(256) void xr_kernel(
    const float* __restrict__ enc, const float* __restrict__ logits,
    float* __restrict__ xr)
{
    int node = blockIdx.x * 256 + threadIdx.x;
    if (node >= BN) return;
    float* o = xr + (size_t)node * XR_PAD;
    const float* e = enc + (size_t)node * ENC_PAD;
    #pragma unroll
    for (int c = 0; c < ENC_DIM; ++c) o[c] = e[c];
    #pragma unroll
    for (int c = 0; c < NCLS_OUT; ++c) o[ENC_DIM + c] = logits[(size_t)node * NCLS_OUT + c];
    #pragma unroll
    for (int c = XR_DIM; c < XR_PAD; ++c) o[c] = 0.0f;
}

// =====================================================================
// Host orchestration
// =====================================================================
extern "C" void kernel_launch(void* const* d_in, const int* in_sizes, int n_in,
                              void* d_out, int out_size, void* d_ws, size_t ws_size,
                              hipStream_t stream) {
    (void)in_sizes; (void)n_in; (void)out_size; (void)ws_size;

    const float* X         = (const float*)d_in[0];
    const float* ffn1_w    = (const float*)d_in[1];
    const float* ffn1_b    = (const float*)d_in[2];
    const float* ffn2_w    = (const float*)d_in[3];
    const float* ffn2_b    = (const float*)d_in[4];
    const float* Rm        = (const float*)d_in[5];
    const float* enc_id_w  = (const float*)d_in[6];
    const float* enc_id_b  = (const float*)d_in[7];
    const float* wt_id     = (const float*)d_in[8];
    const float* bt_id     = (const float*)d_in[9];
    const float* wh_id     = (const float*)d_in[10];
    const float* theta_id  = (const float*)d_in[11];
    const float* dec_id_w  = (const float*)d_in[12];
    const float* dec_id_b  = (const float*)d_in[13];
    const float* out_id_w  = (const float*)d_in[14];
    const float* out_id_b  = (const float*)d_in[15];
    const float* enc_reg_w = (const float*)d_in[16];
    const float* enc_reg_b = (const float*)d_in[17];
    const float* wt_reg    = (const float*)d_in[18];
    const float* bt_reg    = (const float*)d_in[19];
    const float* wh_reg    = (const float*)d_in[20];
    const float* theta_reg = (const float*)d_in[21];
    const float* dec_reg_w = (const float*)d_in[22];
    const float* dec_reg_b = (const float*)d_in[23];
    const float* out_reg_w = (const float*)d_in[24];
    const float* out_reg_b = (const float*)d_in[25];

    float* out_logits = (float*)d_out;                         // [B,N,8]
    float* out_mom    = (float*)d_out + (size_t)BN * NCLS_OUT; // [B,N,3]

    // ---- workspace layout (floats) ----
    float* ws = (float*)d_ws;
    size_t off = 0;
    const size_t SLOT = (size_t)BN * HDIM;   // 10,240,000 floats
    float* enc   = ws + off; off += (size_t)BN * ENC_PAD;
    float* xr    = ws + off; off += (size_t)BN * XR_PAD;
    int*   binid = (int*)(ws + off); off += BN;
    int*   order = (int*)(ws + off); off += BN;
    float* sq    = ws + off; off += BN;
    float* nrm   = ws + off; off += BN;
    int*   esrc  = (int*)(ws + off); off += NEDGE;
    int*   edst  = (int*)(ws + off); off += NEDGE;
    float* eval  = ws + off; off += NEDGE;
    float* P1 = ws + off; off += SLOT;
    float* P2 = ws + off; off += SLOT;
    float* P3 = ws + off; off += SLOT;
    float* P4 = ws + off; off += SLOT;   // Gram matrix uses P4+P5 (exactly 2 slots)
    float* P5 = ws + off; off += SLOT;
    float* P6 = ws + off; off += SLOT;
    (void)P5;

    const dim3 gemmGrid((BN + 127) / 128, HDIM / 64);  // 313 x 4
    const dim3 distGrid(4, 8, BSZ * N_BINS);           // 512x512 tiles per bin
    const int  nodeGrid = (BN + 255) / 256;

    // ---- encode + distance-space MLP ----
    encode_kernel<<<nodeGrid, 256, 0, stream>>>(X, enc);
    gemm_bf16_kernel<<<gemmGrid, 128, 0, stream>>>(enc, ENC_PAD, ENC_DIM, ffn1_w, HDIM,
        ffn1_b, nullptr, P1, HDIM, BN, HDIM, ACT_ELU);
    gemm_bf16_kernel<<<gemmGrid, 128, 0, stream>>>(P1, HDIM, HDIM, ffn2_w, HDIM,
        ffn2_b, nullptr, P2, HDIM, BN, HDIM, ACT_NONE);        // x_dist

    // ---- LSH binning + kNN graph ----
    bin_kernel<<<nodeGrid, 256, 0, stream>>>(P2, Rm, binid);
    sort_kernel<<<BSZ, 256, 0, stream>>>(binid, order);
    parts_kernel<<<BN, 256, 0, stream>>>(P2, order, P3, sq);
    gemm_dist_kernel<<<distGrid, 128, 0, stream>>>(P3, P4);    // Gram in P4..P5
    topk_kernel<<<BSZ * N_BINS * BIN_SIZE, 128, 0, stream>>>(P4, sq, order,
        esrc, edst, eval, nrm);

    // ---- ID branch ----
    gemm_bf16_kernel<<<gemmGrid, 128, 0, stream>>>(enc, ENC_PAD, ENC_DIM, enc_id_w, HDIM,
        enc_id_b, nullptr, P1, HDIM, BN, HDIM, ACT_SELU);                        // h
    gemm_bf16_kernel<<<gemmGrid, 128, 0, stream>>>(P1, HDIM, HDIM, theta_id, HDIM,
        nullptr, nrm, P2, HDIM, BN, HDIM, ACT_NONE);                             // f_hom_pre
    gemm_bf16_kernel<<<gemmGrid, 128, 0, stream>>>(P1, HDIM, HDIM, wh_id, HDIM,
        nullptr, nullptr, P3, HDIM, BN, HDIM, ACT_NONE);                         // f_het
    gemm_bf16_kernel<<<gemmGrid, 128, 0, stream>>>(P1, HDIM, HDIM, wt_id, HDIM,
        bt_id, nullptr, P4, HDIM, BN, HDIM, ACT_NONE);                           // gate pre
    gather_kernel<<<BN, 256, 0, stream>>>(esrc, edst, eval, P2, P5);             // f_agg
    combine_kernel<<<(BN * HDIM) / 256, 256, 0, stream>>>(P4, P5, P3, nrm, P6);  // h2
    gemm_bf16_kernel<<<gemmGrid, 128, 0, stream>>>(P6, HDIM, HDIM, dec_id_w, HDIM,
        dec_id_b, nullptr, P1, HDIM, BN, HDIM, ACT_SELU);                        // h3
    outproj_kernel<<<(BN * NCLS_OUT + 255) / 256, 256, 0, stream>>>(
        P1, out_id_w, out_id_b, out_logits, NCLS_OUT);

    // ---- REG branch ----
    xr_kernel<<<nodeGrid, 256, 0, stream>>>(enc, out_logits, xr);
    gemm_bf16_kernel<<<gemmGrid, 128, 0, stream>>>(xr, XR_PAD, XR_DIM, enc_reg_w, HDIM,
        enc_reg_b, nullptr, P2, HDIM, BN, HDIM, ACT_SELU);                       // g
    gemm_bf16_kernel<<<gemmGrid, 128, 0, stream>>>(P2, HDIM, HDIM, theta_reg, HDIM,
        nullptr, nrm, P3, HDIM, BN, HDIM, ACT_NONE);                             // f_hom_pre
    gemm_bf16_kernel<<<gemmGrid, 128, 0, stream>>>(P2, HDIM, HDIM, wh_reg, HDIM,
        nullptr, nullptr, P4, HDIM, BN, HDIM, ACT_NONE);                         // f_het
    gemm_bf16_kernel<<<gemmGrid, 128, 0, stream>>>(P2, HDIM, HDIM, wt_reg, HDIM,
        bt_reg, nullptr, P5, HDIM, BN, HDIM, ACT_NONE);                          // gate pre
    gather_kernel<<<BN, 256, 0, stream>>>(esrc, edst, eval, P3, P6);             // f_agg
    combine_kernel<<<(BN * HDIM) / 256, 256, 0, stream>>>(P5, P6, P4, nrm, P1);  // g2
    gemm_bf16_kernel<<<gemmGrid, 128, 0, stream>>>(P1, HDIM, HDIM, dec_reg_w, HDIM,
        dec_reg_b, nullptr, P2, HDIM, BN, HDIM, ACT_SELU);                       // g3
    outproj_kernel<<<(BN * NMOM + 255) / 256, 256, 0, stream>>>(
        P2, out_reg_w, out_reg_b, out_mom, NMOM);
}
